// CausalMultiHeadSelfAttention_14860586844419
// MI455X (gfx1250) — compile-verified, matrix operand order checked by reading, not run on hardware
//
#include <hip/hip_runtime.h>

// ---------------------------------------------------------------------------
// Causal MHA for MI455X (gfx1250, wave32, WMMA).
// Compute-bound (~0.2 TFLOP vs ~200MB of HBM traffic @23.3TB/s), so all
// GEMM-shaped work goes through v_wmma_f32_16x16x32_bf16 (f32 accumulate).
// GEMM tiles are staged with double-buffered GLOBAL_LOAD_ASYNC_TO_LDS_B128
// (ASYNCcnt) when the toolchain exposes the builtin.
// ---------------------------------------------------------------------------

typedef unsigned short u16;
typedef __attribute__((ext_vector_type(16))) __bf16 v16bf;
typedef __attribute__((ext_vector_type(8)))  float  v8f;
typedef int v4i_ __attribute__((vector_size(16)));   // int4, matches builtin

#define D_MODEL 1024
#define NHEADS  16
#define DKDIM   64
#define SEQLEN  2048
#define BATCHN  4
#define MTOK    (BATCHN * SEQLEN)   // 8192 tokens

#if __has_builtin(__builtin_amdgcn_global_load_async_to_lds_b128)
#define USE_ASYNC_LDS 1
#endif

// ---------------- scalar bf16 helpers (round-to-nearest-even) --------------
__device__ __forceinline__ u16 f32_to_bf16(float f) {
    union { float f; unsigned u; } v; v.f = f;
    unsigned u = v.u;
    unsigned r = u + 0x7FFFu + ((u >> 16) & 1u);
    return (u16)(r >> 16);
}
__device__ __forceinline__ float bf16_to_f32(u16 h) {
    union { unsigned u; float f; } v; v.u = ((unsigned)h) << 16;
    return v.f;
}

// ---------------- WMMA fragment plumbing -----------------------------------
struct Frag { uint4 lo, hi; };  // 16 bf16 values per lane (8 VGPRs)

union FragCast { Frag f; v16bf v; };

__device__ __forceinline__ v16bf frag_v(const Frag& f) {
    FragCast c; c.f = f; return c.v;
}

__device__ __forceinline__ v8f vzero8() {
    v8f z;
#pragma unroll
    for (int i = 0; i < 8; ++i) z[i] = 0.0f;
    return z;
}

__device__ __forceinline__ v8f wmma_bf16(const Frag& a, const Frag& b, v8f c) {
    // D = A(16x32 bf16) * B(32x16 bf16) + C(16x16 f32)
    return __builtin_amdgcn_wmma_f32_16x16x32_bf16(
        /*neg_a=*/false, frag_v(a), /*neg_b=*/false, frag_v(b),
        /*c_mod=*/(short)0, c, /*reuse_a=*/false, /*reuse_b=*/false);
}

// Load 16 bf16 with the ISA K-packing: {kb..kb+7} and {kb+16..kb+23}
// (kb already folded into p by the caller). p must be 16B aligned.
__device__ __forceinline__ Frag load_frag(const u16* p) {
    Frag f;
    f.lo = *(const uint4*)(p);
    f.hi = *(const uint4*)(p + 16);
    return f;
}

#ifdef USE_ASYNC_LDS
// 16B global -> LDS async copy (no VGPR round trip). The builtin takes
// (int4 as1* src, int4 as3* dst, imm offset, imm cpol); on AMDGPU the low
// 32 bits of a generic LDS pointer are the LDS byte address.
__device__ __forceinline__ void async_copy_b128(const u16* gsrc, u16* ldst) {
    __builtin_amdgcn_global_load_async_to_lds_b128(
        (__attribute__((address_space(1))) v4i_*)(size_t)gsrc,
        (__attribute__((address_space(3))) v4i_*)(unsigned)(size_t)ldst,
        /*offset=*/0, /*cpol=*/0);
}
__device__ __forceinline__ void wait_asynccnt_0() {
#if __has_builtin(__builtin_amdgcn_s_wait_asynccnt)
    __builtin_amdgcn_s_wait_asynccnt(0);
#else
    asm volatile("s_wait_asynccnt 0" ::: "memory");
#endif
}
__device__ __forceinline__ void wait_asynccnt_8() {
#if __has_builtin(__builtin_amdgcn_s_wait_asynccnt)
    __builtin_amdgcn_s_wait_asynccnt(8);
#else
    asm volatile("s_wait_asynccnt 8" ::: "memory");
#endif
}
#endif

// ---------------- elementwise kernels --------------------------------------
__global__ void cvt_f32_to_bf16_kernel(const float* __restrict__ in,
                                       u16* __restrict__ out, int n) {
    int i = blockIdx.x * blockDim.x + threadIdx.x;
    if (i < n) out[i] = f32_to_bf16(in[i]);
}

// In-place interleaved RoPE on a [MTOK, D_MODEL] bf16 tensor. position = s.
__global__ void rope_kernel(u16* __restrict__ t,
                            const float* __restrict__ cosb,
                            const float* __restrict__ sinb) {
    int i = blockIdx.x * blockDim.x + threadIdx.x;       // pair index
    const int NPAIR = MTOK * (D_MODEL / 2);              // 4,194,304
    if (i >= NPAIR) return;
    int tok = i >> 9;              // 512 pairs per token
    int rem = i & 511;
    int h   = rem >> 5;            // head
    int j   = rem & 31;            // rotation pair within head
    int s   = tok & (SEQLEN - 1);  // position
    int base = tok * D_MODEL + h * DKDIM + 2 * j;
    float c  = cosb[s * 32 + j];
    float sn = sinb[s * 32 + j];
    float a = bf16_to_f32(t[base]);
    float b = bf16_to_f32(t[base + 1]);
    t[base]     = f32_to_bf16(a * c - b * sn);
    t[base + 1] = f32_to_bf16(a * sn + b * c);
}

// V [MTOK, D_MODEL] -> Vt [B][H][DK][S] so PV B-fragments load contiguously.
__global__ void transpose_v_kernel(const u16* __restrict__ v,
                                   u16* __restrict__ vt) {
    int i = blockIdx.x * blockDim.x + threadIdx.x;
    if (i >= MTOK * D_MODEL) return;
    int tok = i >> 10;             // / D_MODEL
    int col = i & 1023;
    int b = tok >> 11, s = tok & (SEQLEN - 1);
    int h = col >> 6,  dk = col & 63;
    vt[(((b * NHEADS + h) * DKDIM) + dk) * SEQLEN + s] = v[i];
}

// ---------------- tiled bf16 GEMM: C[M,N] = A[M,K] * W[N,K]^T --------------
// Block tile 128x128, 8 waves each computing 32x64 (2x4 WMMA tiles), BK=64.
#define GK      1024
#define GN      1024
#define LDSK    72              // 64 + 8 pad (ushort elems) to dodge conflicts

template <bool OUTF32>
__global__ __launch_bounds__(256)
void gemm128_bf16_kernel(const u16* __restrict__ A,
                         const u16* __restrict__ W0, const u16* __restrict__ W1,
                         const u16* __restrict__ W2,
                         u16* __restrict__ O0, u16* __restrict__ O1,
                         u16* __restrict__ O2, float* __restrict__ OF) {
    const u16* W;
    u16* OB;
    if (blockIdx.z == 0)      { W = W0; OB = O0; }
    else if (blockIdx.z == 1) { W = W1; OB = O1; }
    else                      { W = W2; OB = O2; }

    const int tid  = threadIdx.x;
    const int wave = tid >> 5;
    const int lane = tid & 31;
    const int lh   = lane & 15;
    const int hi8  = (lane >> 4) * 8;   // K-pack offset == C row offset

    const int moff = (wave & 3) * 32;   // 4 waves along M
    const int noff = (wave >> 2) * 64;  // 2 waves along N
    const int m0 = blockIdx.y * 128;
    const int n0 = blockIdx.x * 128;

    v8f acc[2][4];
#pragma unroll
    for (int mt = 0; mt < 2; ++mt)
#pragma unroll
        for (int nt = 0; nt < 4; ++nt) acc[mt][nt] = vzero8();

#ifdef USE_ASYNC_LDS
    // ---- double-buffered async-to-LDS staging (ASYNCcnt) ----
    __shared__ u16 Xs[2][128 * LDSK];
    __shared__ u16 Ws[2][128 * LDSK];

    // This thread's 4 fixed 8-element segments of the 128x64 tile.
    int segRow[4], segC8[4];
#pragma unroll
    for (int i = 0; i < 4; ++i) {
        int seg = tid + i * 256;
        segRow[i] = seg >> 3;
        segC8[i]  = (seg & 7) * 8;
    }

    // prologue: stage 0 into buffer 0  (8 async b128 copies per thread)
#pragma unroll
    for (int i = 0; i < 4; ++i) {
        async_copy_b128(&A[(m0 + segRow[i]) * GK + segC8[i]],
                        &Xs[0][segRow[i] * LDSK + segC8[i]]);
        async_copy_b128(&W[(n0 + segRow[i]) * GK + segC8[i]],
                        &Ws[0][segRow[i] * LDSK + segC8[i]]);
    }

    int buf = 0;
    for (int kk = 0; kk < GK; kk += 64, buf ^= 1) {
        if (kk + 64 < GK) {
            // issue next stage into the other buffer, then wait only for the
            // current stage (async loads complete in order: <=8 outstanding
            // means the first 8 have landed).
#pragma unroll
            for (int i = 0; i < 4; ++i) {
                async_copy_b128(&A[(m0 + segRow[i]) * GK + kk + 64 + segC8[i]],
                                &Xs[buf ^ 1][segRow[i] * LDSK + segC8[i]]);
                async_copy_b128(&W[(n0 + segRow[i]) * GK + kk + 64 + segC8[i]],
                                &Ws[buf ^ 1][segRow[i] * LDSK + segC8[i]]);
            }
            wait_asynccnt_8();
        } else {
            wait_asynccnt_0();
        }
        __syncthreads();

#pragma unroll
        for (int ks = 0; ks < 64; ks += 32) {
            Frag af[2];
#pragma unroll
            for (int mt = 0; mt < 2; ++mt)
                af[mt] = load_frag(&Xs[buf][(moff + mt * 16 + lh) * LDSK + ks + hi8]);
#pragma unroll
            for (int nt = 0; nt < 4; ++nt) {
                Frag bf = load_frag(&Ws[buf][(noff + nt * 16 + lh) * LDSK + ks + hi8]);
#pragma unroll
                for (int mt = 0; mt < 2; ++mt)
                    acc[mt][nt] = wmma_bf16(af[mt], bf, acc[mt][nt]);
            }
        }
        __syncthreads();  // all waves done reading buf before it is refilled
    }
#else
    // ---- fallback: synchronous staging through VGPRs ----
    __shared__ u16 Xs[128 * LDSK];
    __shared__ u16 Ws[128 * LDSK];

    for (int kk = 0; kk < GK; kk += 64) {
        __syncthreads();
#pragma unroll
        for (int i = 0; i < 4; ++i) {
            int seg = tid + i * 256;
            int row = seg >> 3;
            int c8  = (seg & 7) * 8;
            *(uint4*)&Xs[row * LDSK + c8] =
                *(const uint4*)&A[(m0 + row) * GK + kk + c8];
            *(uint4*)&Ws[row * LDSK + c8] =
                *(const uint4*)&W[(n0 + row) * GK + kk + c8];
        }
        if (kk + 64 < GK) {
            __builtin_prefetch(&A[(m0 + (tid >> 1)) * GK + kk + 64 + (tid & 1) * 32], 0, 1);
            __builtin_prefetch(&W[(n0 + (tid >> 1)) * GK + kk + 64 + (tid & 1) * 32], 0, 1);
        }
        __syncthreads();

#pragma unroll
        for (int ks = 0; ks < 64; ks += 32) {
            Frag af[2];
#pragma unroll
            for (int mt = 0; mt < 2; ++mt)
                af[mt] = load_frag(&Xs[(moff + mt * 16 + lh) * LDSK + ks + hi8]);
#pragma unroll
            for (int nt = 0; nt < 4; ++nt) {
                Frag bf = load_frag(&Ws[(noff + nt * 16 + lh) * LDSK + ks + hi8]);
#pragma unroll
                for (int mt = 0; mt < 2; ++mt)
                    acc[mt][nt] = wmma_bf16(af[mt], bf, acc[mt][nt]);
            }
        }
    }
#endif

    // C/D layout: lane holds column lh, VGPR r holds row r + hi8.
#pragma unroll
    for (int mt = 0; mt < 2; ++mt)
#pragma unroll
        for (int nt = 0; nt < 4; ++nt)
#pragma unroll
            for (int r = 0; r < 8; ++r) {
                int row = m0 + moff + mt * 16 + r + hi8;
                int col = n0 + noff + nt * 16 + lh;
                if (OUTF32) OF[row * GN + col] = acc[mt][nt][r];
                else        OB[row * GN + col] = f32_to_bf16(acc[mt][nt][r]);
            }
}

// ---------------- causal flash attention (per-wave 16-query tile) ----------
#define PLD 72   // LDS row stride for the P bounce buffer

__global__ __launch_bounds__(256)
void flash_attn_kernel(const u16* __restrict__ Qb, const u16* __restrict__ Kb,
                       const u16* __restrict__ Vt, u16* __restrict__ ctx) {
    const int wave = threadIdx.x >> 5;
    const int lane = threadIdx.x & 31;
    const int lh   = lane & 15;
    const int hi8  = (lane >> 4) * 8;

    const int bh = blockIdx.y;           // 0..63
    const int b  = bh >> 4;
    const int h  = bh & 15;
    const int q0 = blockIdx.x * 128 + wave * 16;

    __shared__ u16 Pl[8][16 * PLD];
    u16* P = Pl[wave];

    // Q A-fragments (16 rows x 64 k) — contiguous global loads.
    Frag qf[2];
    {
        int tok = b * SEQLEN + q0 + lh;
#pragma unroll
        for (int c = 0; c < 2; ++c)
            qf[c] = load_frag(&Qb[tok * D_MODEL + h * DKDIM + c * 32 + hi8]);
    }

    v8f acc[4];
#pragma unroll
    for (int n = 0; n < 4; ++n) acc[n] = vzero8();
    float m[8], l[8];
#pragma unroll
    for (int r = 0; r < 8; ++r) { m[r] = -1e30f; l[r] = 0.0f; }

    const float scale = 0.125f;                 // 1/sqrt(64)
    const int nkb = (q0 + 16 + 63) >> 6;        // causally-bounded 64-key blocks

    for (int kblk = 0; kblk < nkb; ++kblk) {
        const int key0 = kblk * 64;

        // S = Q K^T : 4 score tiles of 16x16
        v8f st[4];
#pragma unroll
        for (int t = 0; t < 4; ++t) {
            int ktok = b * SEQLEN + key0 + t * 16 + lh;   // B lane = key column
            Frag kf0 = load_frag(&Kb[ktok * D_MODEL + h * DKDIM + 0  + hi8]);
            Frag kf1 = load_frag(&Kb[ktok * D_MODEL + h * DKDIM + 32 + hi8]);
            v8f zc = vzero8();
            st[t] = wmma_bf16(qf[0], kf0, zc);
            st[t] = wmma_bf16(qf[1], kf1, st[t]);
        }

        // scale + causal mask + row max
        float rowm[8];
#pragma unroll
        for (int r = 0; r < 8; ++r) rowm[r] = -1e30f;
#pragma unroll
        for (int t = 0; t < 4; ++t) {
            int key = key0 + t * 16 + lh;
#pragma unroll
            for (int r = 0; r < 8; ++r) {
                int qrow = q0 + r + hi8;
                float v = st[t][r] * scale;
                if (key > qrow) v = -1e30f;
                st[t][r] = v;
                rowm[r] = fmaxf(rowm[r], v);
            }
        }
#pragma unroll
        for (int r = 0; r < 8; ++r)
#pragma unroll
            for (int mk = 1; mk < 16; mk <<= 1)
                rowm[r] = fmaxf(rowm[r], __shfl_xor(rowm[r], mk, 32));

        // online softmax update + P (bf16) through LDS for layout change
        float alpha[8], rowsum[8];
#pragma unroll
        for (int r = 0; r < 8; ++r) {
            float mnew = fmaxf(m[r], rowm[r]);
            alpha[r] = __expf(m[r] - mnew);
            m[r] = mnew;
            rowsum[r] = 0.0f;
        }
#pragma unroll
        for (int t = 0; t < 4; ++t)
#pragma unroll
            for (int r = 0; r < 8; ++r) {
                float p = __expf(st[t][r] - m[r]);
                rowsum[r] += p;
                P[(r + hi8) * PLD + t * 16 + lh] = f32_to_bf16(p);
            }
#pragma unroll
        for (int r = 0; r < 8; ++r) {
#pragma unroll
            for (int mk = 1; mk < 16; mk <<= 1)
                rowsum[r] += __shfl_xor(rowsum[r], mk, 32);
            l[r] = l[r] * alpha[r] + rowsum[r];
        }
#pragma unroll
        for (int n = 0; n < 4; ++n)
#pragma unroll
            for (int r = 0; r < 8; ++r) acc[n][r] *= alpha[r];

        // wave-private LDS: only need DS counter drain before re-reading
        asm volatile("s_wait_dscnt 0" ::: "memory");

        Frag pf[2];
#pragma unroll
        for (int c = 0; c < 2; ++c)
            pf[c] = load_frag(&P[lh * PLD + c * 32 + hi8]);

        // O += P * V  (B-fragments from V^T: contiguous along keys)
#pragma unroll
        for (int n = 0; n < 4; ++n)
#pragma unroll
            for (int c = 0; c < 2; ++c) {
                const u16* vp = &Vt[((bh * DKDIM) + n * 16 + lh) * SEQLEN +
                                    key0 + c * 32 + hi8];
                Frag vf = load_frag(vp);
                acc[n] = wmma_bf16(pf[c], vf, acc[n]);
            }
    }

    // epilogue: normalize and store context (bf16) for the Wo GEMM
    float invl[8];
#pragma unroll
    for (int r = 0; r < 8; ++r) invl[r] = 1.0f / l[r];
#pragma unroll
    for (int n = 0; n < 4; ++n)
#pragma unroll
        for (int r = 0; r < 8; ++r) {
            int row = q0 + r + hi8;
            ctx[(b * SEQLEN + row) * D_MODEL + h * DKDIM + n * 16 + lh] =
                f32_to_bf16(acc[n][r] * invl[r]);
        }
}

// ---------------------------------------------------------------------------
extern "C" void kernel_launch(void* const* d_in, const int* in_sizes, int n_in,
                              void* d_out, int out_size, void* d_ws, size_t ws_size,
                              hipStream_t stream) {
    (void)in_sizes; (void)n_in; (void)out_size; (void)ws_size;

    const float* x    = (const float*)d_in[0];
    // d_in[1] = pos_ids: positions are just arange(S); handled implicitly.
    const float* Wq   = (const float*)d_in[2];
    const float* Wk   = (const float*)d_in[3];
    const float* Wv   = (const float*)d_in[4];
    const float* Wo   = (const float*)d_in[5];
    const float* cosb = (const float*)d_in[6];
    const float* sinb = (const float*)d_in[7];
    float* out = (float*)d_out;

    // workspace layout (bytes); ctx reuses the x_bf region after projections
    char* ws = (char*)d_ws;
    const size_t SZ_X  = (size_t)MTOK * D_MODEL * 2;      // 16 MB
    const size_t SZ_W  = (size_t)D_MODEL * D_MODEL * 2;   // 2 MB
    u16* xbf  = (u16*)(ws);
    u16* wqbf = (u16*)(ws + SZ_X);
    u16* wkbf = (u16*)(ws + SZ_X + SZ_W);
    u16* wvbf = (u16*)(ws + SZ_X + 2 * SZ_W);
    u16* wobf = (u16*)(ws + SZ_X + 3 * SZ_W);
    u16* qbf  = (u16*)(ws + SZ_X + 4 * SZ_W);
    u16* kbf  = (u16*)(ws + 2 * SZ_X + 4 * SZ_W);
    u16* vbf  = (u16*)(ws + 3 * SZ_X + 4 * SZ_W);
    u16* vt   = (u16*)(ws + 4 * SZ_X + 4 * SZ_W);
    u16* ctx  = xbf;  // x no longer needed after the QKV projections

    const int NX = MTOK * D_MODEL;       // 8,388,608
    const int NW = D_MODEL * D_MODEL;    // 1,048,576

    // 1) f32 -> bf16 conversions
    cvt_f32_to_bf16_kernel<<<(NX + 255) / 256, 256, 0, stream>>>(x,  xbf,  NX);
    cvt_f32_to_bf16_kernel<<<(NW + 255) / 256, 256, 0, stream>>>(Wq, wqbf, NW);
    cvt_f32_to_bf16_kernel<<<(NW + 255) / 256, 256, 0, stream>>>(Wk, wkbf, NW);
    cvt_f32_to_bf16_kernel<<<(NW + 255) / 256, 256, 0, stream>>>(Wv, wvbf, NW);
    cvt_f32_to_bf16_kernel<<<(NW + 255) / 256, 256, 0, stream>>>(Wo, wobf, NW);

    // 2) Q/K/V projections: one grid, z selects the weight/output
    dim3 gemmGrid(D_MODEL / 128, MTOK / 128, 3);
    gemm128_bf16_kernel<false><<<gemmGrid, 256, 0, stream>>>(
        xbf, wqbf, wkbf, wvbf, qbf, kbf, vbf, nullptr);

    // 3) RoPE on Q and K (in place, bf16)
    const int NPAIR = MTOK * (D_MODEL / 2);
    rope_kernel<<<(NPAIR + 255) / 256, 256, 0, stream>>>(qbf, cosb, sinb);
    rope_kernel<<<(NPAIR + 255) / 256, 256, 0, stream>>>(kbf, cosb, sinb);

    // 4) V -> V^T per (b,h)
    transpose_v_kernel<<<(NX + 255) / 256, 256, 0, stream>>>(vbf, vt);

    // 5) causal flash attention
    dim3 attnGrid(SEQLEN / 128, BATCHN * NHEADS, 1);
    flash_attn_kernel<<<attnGrid, 256, 0, stream>>>(qbf, kbf, vt, ctx);

    // 6) output projection -> f32 d_out
    dim3 outGrid(D_MODEL / 128, MTOK / 128, 1);
    gemm128_bf16_kernel<true><<<outGrid, 256, 0, stream>>>(
        ctx, wobf, wobf, wobf, nullptr, nullptr, nullptr, out);
}
